// Attention_987842478290
// MI455X (gfx1250) — matrix operand order repaired and checked by reading, not compiled, not run on hardware
//
#include <hip/hip_runtime.h>

// ---------------------------------------------------------------------------
// MHA for MI455X (gfx1250): bf16 WMMA everywhere, fp32 accumulation.
// ---------------------------------------------------------------------------

typedef __attribute__((ext_vector_type(16))) __bf16 v16bf;
typedef __attribute__((ext_vector_type(8)))  __bf16 v8bf;
typedef __attribute__((ext_vector_type(2)))  __bf16 v2bf;
typedef __attribute__((ext_vector_type(8)))  float  v8f;

#define EMBED 1024
#define HEADS 16
#define HDIM  64
#define BATCH 2
#define SEQ   2048
#define ROWS  (BATCH * SEQ)   // 4096

// Native fptrunc -> hardware bf16 convert (RNE) on gfx1250.
static __device__ __forceinline__ __bf16 f2bf(float f) { return (__bf16)f; }

static __device__ __forceinline__ v8f wmma_bf16(v16bf a, v16bf b, v8f c) {
  // D = A(16x32) * B(32x16) + C, f32 accumulate
  return __builtin_amdgcn_wmma_f32_16x16x32_bf16(
      /*neg_a=*/false, a, /*neg_b=*/false, b,
      /*c_mod=*/(short)0, c, /*reuse_a=*/false, /*reuse_b=*/false);
}

// ---------------------------------------------------------------------------
// Kernel 1: W[K][N] fp32  ->  Wt[N][K] bf16 (tiled transpose, coalesced)
// ---------------------------------------------------------------------------
__global__ __launch_bounds__(1024) void transpose_to_bf16(
    const float* __restrict__ src, __bf16* __restrict__ dst) {
  __shared__ float tile[32][33];
  int x = blockIdx.x * 32 + threadIdx.x;   // N index on read
  int y = blockIdx.y * 32 + threadIdx.y;   // K index on read
  tile[threadIdx.y][threadIdx.x] = src[y * EMBED + x];
  __syncthreads();
  int n = blockIdx.x * 32 + threadIdx.y;
  int k = blockIdx.y * 32 + threadIdx.x;
  dst[n * EMBED + k] = f2bf(tile[threadIdx.x][threadIdx.y]);
}

// ---------------------------------------------------------------------------
// Kernel 2: C[M,1024] = A[M,1024] @ Wt^T + bias, WMMA bf16.
//   mode 0: out bf16 head-split [bh][n][64]              (Q, K projections)
//   mode 1: out bf16 head-split transposed [bh][64][n]   (V projection)
//   mode 2: out fp32 row-major                           (final projection)
// Block: 256 threads (8 waves). Block tile 128x256; wave tile 64x64
// (16 accumulators -> 16 v_wmma per 32-wide K-step).
// ---------------------------------------------------------------------------
__global__ __launch_bounds__(256) void gemm_wmma(
    const float*  __restrict__ A32, const __bf16* __restrict__ A16,
    const __bf16* __restrict__ Wt,  const float*  __restrict__ bias,
    float* __restrict__ outF, __bf16* __restrict__ outB, int mode) {
  __shared__ __align__(16) __bf16 As[128][40];   // 32 valid K + pad
  __shared__ __align__(16) __bf16 Bs[256][40];

  const int tid   = threadIdx.x;
  const int lane  = tid & 31;
  const int wave  = tid >> 5;
  const int half  = lane >> 4;
  const int l15   = lane & 15;
  const int waveM = wave & 1;    // 2 M-groups of 64
  const int waveN = wave >> 1;   // 4 N-groups of 64
  const int blockM = blockIdx.x * 128;
  const int blockN = blockIdx.y * 256;

  const int ldrow = tid >> 1;          // 0..127
  const int ldseg = (tid & 1) * 16;    // 0 or 16

  const v8f vzero = {0.f,0.f,0.f,0.f,0.f,0.f,0.f,0.f};
  v8f acc[4][4];
#pragma unroll
  for (int mt = 0; mt < 4; mt++)
#pragma unroll
    for (int nt = 0; nt < 4; nt++) acc[mt][nt] = vzero;

  for (int ks = 0; ks < EMBED; ks += 32) {
    // ---- stage A tile (128x32) into LDS as bf16 ----
    {
      int grow = blockM + ldrow;
      if (A32) {
        const float4* p = (const float4*)(A32 + (size_t)grow * EMBED + ks + ldseg);
        float tmp[16];
        ((float4*)tmp)[0] = p[0]; ((float4*)tmp)[1] = p[1];
        ((float4*)tmp)[2] = p[2]; ((float4*)tmp)[3] = p[3];
        v8bf h0, h1;
#pragma unroll
        for (int c = 0; c < 8; c++) { h0[c] = f2bf(tmp[c]); h1[c] = f2bf(tmp[8 + c]); }
        *(v8bf*)&As[ldrow][ldseg]     = h0;
        *(v8bf*)&As[ldrow][ldseg + 8] = h1;
        if (ks + 32 < EMBED)
          __builtin_prefetch(A32 + (size_t)grow * EMBED + ks + 32 + ldseg, 0, 3);
      } else {
        const v8bf* p = (const v8bf*)(A16 + (size_t)grow * EMBED + ks + ldseg);
        *(v8bf*)&As[ldrow][ldseg]     = p[0];
        *(v8bf*)&As[ldrow][ldseg + 8] = p[1];
        if (ks + 32 < EMBED)
          __builtin_prefetch(A16 + (size_t)grow * EMBED + ks + 32 + ldseg, 0, 3);
      }
      // ---- stage Wt tile (256 output cols x 32 K): 2 rows per thread ----
#pragma unroll
      for (int rr = 0; rr < 2; rr++) {
        int br = ldrow + rr * 128;
        const v8bf* q = (const v8bf*)(Wt + (size_t)(blockN + br) * EMBED + ks + ldseg);
        *(v8bf*)&Bs[br][ldseg]     = q[0];
        *(v8bf*)&Bs[br][ldseg + 8] = q[1];
        if (ks + 32 < EMBED)
          __builtin_prefetch(Wt + (size_t)(blockN + br) * EMBED + ks + 32 + ldseg, 0, 3);
      }
    }
    __syncthreads();

    // ---- build fragments (exact CDNA5 K permutation) ----
    v16bf aF[4], bF[4];
#pragma unroll
    for (int mt = 0; mt < 4; mt++) {
      int r = waveM * 64 + mt * 16 + l15;
      v8bf lo = *(const v8bf*)&As[r][half * 8];        // K: h? {0..7} : {8..15}
      v8bf hi = *(const v8bf*)&As[r][half * 8 + 16];   // K: h? {16..23} : {24..31}
#pragma unroll
      for (int c = 0; c < 8; c++) { aF[mt][c] = lo[c]; aF[mt][8 + c] = hi[c]; }
    }
#pragma unroll
    for (int nt = 0; nt < 4; nt++) {
      int r = waveN * 64 + nt * 16 + l15;
      v8bf lo = *(const v8bf*)&Bs[r][half * 8];        // K: half*8 + 0..7
      v8bf hi = *(const v8bf*)&Bs[r][half * 8 + 16];   // K: 16 + half*8 + 0..7
#pragma unroll
      for (int c = 0; c < 8; c++) { bF[nt][c] = lo[c]; bF[nt][8 + c] = hi[c]; }
    }

#pragma unroll
    for (int mt = 0; mt < 4; mt++)
#pragma unroll
      for (int nt = 0; nt < 4; nt++)
        acc[mt][nt] = wmma_bf16(aF[mt], bF[nt], acc[mt][nt]);
    __syncthreads();
  }

  // ---- epilogue: bias + layout-specific store ----
#pragma unroll
  for (int nt = 0; nt < 4; nt++) {
    int colg = blockN + waveN * 64 + nt * 16 + l15;
    float bv = bias[colg];
    int h = colg >> 6, d = colg & 63;
#pragma unroll
    for (int mt = 0; mt < 4; mt++) {
      int rbase = blockM + waveM * 64 + mt * 16 + half * 8;
      if (mode == 2) {
#pragma unroll
        for (int i = 0; i < 8; i++)
          outF[(size_t)(rbase + i) * EMBED + colg] = acc[mt][nt][i] + bv;
      } else if (mode == 0) {
#pragma unroll
        for (int i = 0; i < 8; i++) {
          int rowg = rbase + i;
          int b = rowg >> 11, nr = rowg & (SEQ - 1);
          size_t idx = ((size_t)((b * HEADS + h) * SEQ + nr)) * HDIM + d;
          outB[idx] = f2bf(acc[mt][nt][i] + bv);
        }
      } else {
        // mode 1: consecutive i -> consecutive token index: packed dword stores
        int b = rbase >> 11, nr = rbase & (SEQ - 1);
        size_t base = ((size_t)((b * HEADS + h) * HDIM + d)) * SEQ + nr;
#pragma unroll
        for (int i = 0; i < 8; i += 2) {
          v2bf pkt = { f2bf(acc[mt][nt][i] + bv), f2bf(acc[mt][nt][i + 1] + bv) };
          *(v2bf*)(outB + base + i) = pkt;
        }
      }
    }
  }
}

// ---------------------------------------------------------------------------
// Kernel 3: flash attention, transposed formulation.
//   S^T = K * Q^T  (softmax axis m lands on in-lane VGPR axis of C layout)
//   O^T = V^T * P^T
// One wave per 16 queries; block = 8 waves; grid = (b*h, SEQ/128).
// ---------------------------------------------------------------------------
__global__ __launch_bounds__(256) void flash_attn(
    const __bf16* __restrict__ Qh, const __bf16* __restrict__ Kh,
    const __bf16* __restrict__ Vt, __bf16* __restrict__ O) {
  const int tid  = threadIdx.x;
  const int lane = tid & 31;
  const int wave = tid >> 5;
  const int half = lane >> 4;
  const int l15  = lane & 15;
  const int bh   = blockIdx.x;                    // 0..31
  const int q0   = (blockIdx.y * 8 + wave) * 16;  // query tile base
  const int hh   = bh & (HEADS - 1);
  const int rowg = (bh >> 4) * SEQ + q0 + l15;    // this lane's query row

  // Q as B-fragments (column j = query q0+j; K halves split across lane halves)
  const __bf16* qrow = Qh + ((size_t)bh * SEQ + q0 + l15) * HDIM;
  v16bf bq0 = *(const v16bf*)(qrow + half * 16);        // d 0..31
  v16bf bq1 = *(const v16bf*)(qrow + 32 + half * 16);   // d 32..63

  const v8f vzero = {0.f,0.f,0.f,0.f,0.f,0.f,0.f,0.f};
  v8f accO[4];
#pragma unroll
  for (int t = 0; t < 4; t++) accO[t] = vzero;
  float m_run = -1e30f, l_run = 0.f;

  for (int m0 = 0; m0 < SEQ; m0 += 32) {
    // ---- S^T tiles for m rows [m0,m0+16) and [m0+16,m0+32) ----
    v8f st[2];
#pragma unroll
    for (int t = 0; t < 2; t++) {
      const __bf16* krow = Kh + ((size_t)bh * SEQ + m0 + t * 16 + l15) * HDIM;
      v16bf a0, a1;
      {
        v8bf lo = *(const v8bf*)(krow + half * 16);
        v8bf hi = *(const v8bf*)(krow + half * 16 + 8);
#pragma unroll
        for (int c = 0; c < 8; c++) { a0[c] = lo[c]; a0[8 + c] = hi[c]; }
        lo = *(const v8bf*)(krow + 32 + half * 16);
        hi = *(const v8bf*)(krow + 32 + half * 16 + 8);
#pragma unroll
        for (int c = 0; c < 8; c++) { a1[c] = lo[c]; a1[8 + c] = hi[c]; }
      }
      v8f c = vzero;
      c = wmma_bf16(a0, bq0, c);
      c = wmma_bf16(a1, bq1, c);
      st[t] = c * 0.125f;   // 1/sqrt(64)
    }

    // ---- online softmax (lane pair j / j+16 shares column j) ----
    float cm = -1e30f;
#pragma unroll
    for (int i = 0; i < 8; i++) cm = fmaxf(cm, fmaxf(st[0][i], st[1][i]));
    cm = fmaxf(cm, __shfl_xor(cm, 16, 32));
    float nm = fmaxf(m_run, cm);
    float r  = __expf(m_run - nm);
    m_run = nm;

    float e0[8], e1[8];
#pragma unroll
    for (int i = 0; i < 8; i++) {
      e0[i] = __expf(st[0][i] - nm);
      e1[i] = __expf(st[1][i] - nm);
    }
    // Rebuild P^T B-fragment: lane<16 needs m {0..15}, lane>=16 needs m {16..31}
    v16bf pf;
    float ls = 0.f;
#pragma unroll
    for (int i = 0; i < 8; i++) {
      float o0 = __shfl_xor(e0[i], 16, 32);
      float o1 = __shfl_xor(e1[i], 16, 32);
      float a = (half == 0) ? e0[i] : o1;   // K slots 0..7 of this lane
      float b = (half == 0) ? o0 : e1[i];   // K slots 8..15
      pf[i] = f2bf(a);
      pf[8 + i] = f2bf(b);
      ls += a + b;
    }
    l_run = l_run * r + ls;

#pragma unroll
    for (int t = 0; t < 4; t++) accO[t] = accO[t] * r;

    // ---- O^T += Vt_tile(16 d-rows x 32 m) * P^T ----
#pragma unroll
    for (int t = 0; t < 4; t++) {
      const __bf16* vrow = Vt + ((size_t)bh * HDIM + t * 16 + l15) * SEQ + m0;
      v16bf av;
      v8bf lo = *(const v8bf*)(vrow + half * 16);
      v8bf hi = *(const v8bf*)(vrow + half * 16 + 8);
#pragma unroll
      for (int c = 0; c < 8; c++) { av[c] = lo[c]; av[8 + c] = hi[c]; }
      accO[t] = wmma_bf16(av, pf, accO[t]);
    }
  }

  float ltot = l_run + __shfl_xor(l_run, 16, 32);
  float inv  = 1.f / ltot;
  // contiguous d within lane -> packed dword stores
#pragma unroll
  for (int t = 0; t < 4; t++) {
    int col = hh * HDIM + t * 16 + half * 8;
#pragma unroll
    for (int i = 0; i < 8; i += 2) {
      v2bf pkt = { f2bf(accO[t][i] * inv), f2bf(accO[t][i + 1] * inv) };
      *(v2bf*)(O + (size_t)rowg * EMBED + col + i) = pkt;
    }
  }
}

// ---------------------------------------------------------------------------
extern "C" void kernel_launch(void* const* d_in, const int* in_sizes, int n_in,
                              void* d_out, int out_size, void* d_ws, size_t ws_size,
                              hipStream_t stream) {
  (void)in_sizes; (void)n_in; (void)out_size; (void)ws_size;
  const float* q  = (const float*)d_in[0];
  const float* k  = (const float*)d_in[1];
  const float* v  = (const float*)d_in[2];
  const float* Wq = (const float*)d_in[3];
  const float* bq = (const float*)d_in[4];
  const float* Wk = (const float*)d_in[5];
  const float* bk = (const float*)d_in[6];
  const float* Wv = (const float*)d_in[7];
  const float* bv = (const float*)d_in[8];
  const float* Wo = (const float*)d_in[9];
  const float* bo = (const float*)d_in[10];
  float* out = (float*)d_out;

  // workspace layout (bytes): 4x Wt (2MB) | Qh 8MB | Kh 8MB | Vt 8MB | O 8MB
  char* ws = (char*)d_ws;
  const size_t WB = (size_t)EMBED * EMBED * sizeof(__bf16);   // 2 MB
  const size_t AB = (size_t)ROWS * EMBED * sizeof(__bf16);    // 8 MB
  __bf16* Wt0 = (__bf16*)(ws + 0 * WB);
  __bf16* Wt1 = (__bf16*)(ws + 1 * WB);
  __bf16* Wt2 = (__bf16*)(ws + 2 * WB);
  __bf16* Wt3 = (__bf16*)(ws + 3 * WB);
  __bf16* Qh  = (__bf16*)(ws + 4 * WB);
  __bf16* Kh  = (__bf16*)(ws + 4 * WB + 1 * AB);
  __bf16* Vt  = (__bf16*)(ws + 4 * WB + 2 * AB);
  __bf16* Ob  = (__bf16*)(ws + 4 * WB + 3 * AB);

  dim3 tB(32, 32), tG(EMBED / 32, EMBED / 32);
  transpose_to_bf16<<<tG, tB, 0, stream>>>(Wq, Wt0);
  transpose_to_bf16<<<tG, tB, 0, stream>>>(Wk, Wt1);
  transpose_to_bf16<<<tG, tB, 0, stream>>>(Wv, Wt2);
  transpose_to_bf16<<<tG, tB, 0, stream>>>(Wo, Wt3);

  dim3 gG(ROWS / 128, EMBED / 256);
  gemm_wmma<<<gG, 256, 0, stream>>>(q, nullptr, Wt0, bq, nullptr, Qh, 0);
  gemm_wmma<<<gG, 256, 0, stream>>>(k, nullptr, Wt1, bk, nullptr, Kh, 0);
  gemm_wmma<<<gG, 256, 0, stream>>>(v, nullptr, Wt2, bv, nullptr, Vt, 1);

  flash_attn<<<dim3(BATCH * HEADS, SEQ / 128), 256, 0, stream>>>(Qh, Kh, Vt, Ob);

  gemm_wmma<<<gG, 256, 0, stream>>>(nullptr, Ob, Wt3, bo, out, nullptr, 2);
}
